// LinearCrossNetwork_90795608638308
// MI455X (gfx1250) — matrix operand order repaired and checked by reading, not compiled, not run on hardware
//
#include <hip/hip_runtime.h>
#include <hip/hip_bf16.h>
#include <math.h>

// ---------------- problem constants (match reference) ----------------
constexpr int Bn = 16384;   // batch
constexpr int Dd = 1024;    // feature dim
constexpr int Hh = 512;     // half dim
constexpr int Ll = 3;       // layers
constexpr float EPS = 1e-5f;

typedef __bf16 bf16_t;
typedef __attribute__((ext_vector_type(16))) __bf16 v16bf;
typedef __attribute__((ext_vector_type(8)))  float  v8f;

// =====================================================================
// Fragment-major swizzled operand layouts (so every GEMM load is a
// contiguous 32B-per-lane, fully coalesced v16bf load):
//   A (activations, M x K):  blk = (m0/16)*(K/32) + (k0/32)
//     elem o = blk*512 + lane*16 + e  holds  A[m0+(lane&15)][kA(e,lane)]
//     kA = k0 + (e<8?0:16) + 8*(lane>>4) + (e&7)        (ISA 16-bit A frag)
//   B (weights stored [N x K], used as B = W^T):
//     elem o = blk*512 + lane*16 + e  holds  W[n0+(lane&15)][kA(e,lane)]
// =====================================================================

template<int K, bool AFFINE>
__global__ void swz_a_split(const float* __restrict__ src,
                            const float* __restrict__ s, const float* __restrict__ t,
                            bf16_t* __restrict__ hi, bf16_t* __restrict__ lo,
                            int nElem) {
    int o = blockIdx.x * blockDim.x + threadIdx.x;
    int stride = gridDim.x * blockDim.x;
    constexpr int KT = K / 32;
    for (; o < nElem; o += stride) {
        int e    = o & 15;
        int lane = (o >> 4) & 31;
        int blk  = o >> 9;
        int kt   = blk % KT;
        int mt   = blk / KT;
        int m    = mt * 16 + (lane & 15);
        int kk   = kt * 32 + ((e < 8) ? 0 : 16) + 8 * (lane >> 4) + (e & 7);
        float v = src[(size_t)m * K + kk];
        if (AFFINE) v = v * s[kk] + t[kk];
        __bf16 h = (__bf16)v;
        hi[o] = h;
        lo[o] = (__bf16)(v - (float)h);
    }
}

template<int K>
__global__ void swz_b_split(const float* __restrict__ src /* [N x K] */,
                            bf16_t* __restrict__ hi, bf16_t* __restrict__ lo,
                            int nElem) {
    int o = blockIdx.x * blockDim.x + threadIdx.x;
    int stride = gridDim.x * blockDim.x;
    constexpr int KT = K / 32;
    for (; o < nElem; o += stride) {
        int e    = o & 15;
        int lane = (o >> 4) & 31;
        int blk  = o >> 9;
        int kt   = blk % KT;
        int nt   = blk / KT;
        int n    = nt * 16 + (lane & 15);
        int kk   = kt * 32 + ((e < 8) ? 0 : 16) + 8 * (lane >> 4) + (e & 7);
        float v = src[(size_t)n * K + kk];
        __bf16 h = (__bf16)v;
        hi[o] = h;
        lo[o] = (__bf16)(v - (float)h);
    }
}

// =====================================================================
// Split-bf16 GEMM on swizzled operands: C[Bn x N] = A * W^T, fp32 acc,
// A~=Ahi+Alo, W~=Whi+Wlo, C += Ahi*Whi + Ahi*Wlo + Alo*Whi.
// Wave tile 32x64 (2 M-subtiles x 4 N-subtiles, 8 accumulators);
// block = 8 waves (2x4) -> 64x256 tile.  All loads contiguous 32B/lane.
// =====================================================================
template<int K>
__global__ __launch_bounds__(256) void gemm_swz(
        const bf16_t* __restrict__ Ahi, const bf16_t* __restrict__ Alo,
        const bf16_t* __restrict__ Bhi, const bf16_t* __restrict__ Blo,
        float* __restrict__ C, int N) {
    constexpr int KT = K / 32;
    const int lane = threadIdx.x & 31;
    const int wave = threadIdx.x >> 5;
    const int wm = wave >> 2;            // 0..1
    const int wn = wave & 3;             // 0..3
    const int rowT = blockIdx.y * 4 + wm * 2;    // 16-row tile index (wave: +0,+1)
    const int colT = blockIdx.x * 16 + wn * 4;   // 16-col tile index (wave: +0..+3)
    const int half = lane >> 4;
    const int l15  = lane & 15;

    v8f acc[2][4] = {};

    const bf16_t* aH[2];
    const bf16_t* aL[2];
#pragma unroll
    for (int ms = 0; ms < 2; ++ms) {
        size_t base = ((size_t)(rowT + ms) * KT) * 512 + lane * 16;
        aH[ms] = Ahi + base;
        aL[ms] = Alo + base;
    }
    const bf16_t* bH[4];
    const bf16_t* bL[4];
#pragma unroll
    for (int ns = 0; ns < 4; ++ns) {
        size_t base = ((size_t)(colT + ns) * KT) * 512 + lane * 16;
        bH[ns] = Bhi + base;
        bL[ns] = Blo + base;
    }

    for (int kt = 0; kt < KT; ++kt) {
        const size_t ko = (size_t)kt * 512;
        v16bf ah0 = *(const v16bf*)(aH[0] + ko);
        v16bf al0 = *(const v16bf*)(aL[0] + ko);
        v16bf ah1 = *(const v16bf*)(aH[1] + ko);
        v16bf al1 = *(const v16bf*)(aL[1] + ko);
#pragma unroll
        for (int ns = 0; ns < 4; ++ns) {
            v16bf bh = *(const v16bf*)(bH[ns] + ko);
            v16bf bl = *(const v16bf*)(bL[ns] + ko);
            acc[0][ns] = __builtin_amdgcn_wmma_f32_16x16x32_bf16(false, ah0, false, bh,
                                                                 (short)0, acc[0][ns], false, false);
            acc[0][ns] = __builtin_amdgcn_wmma_f32_16x16x32_bf16(false, ah0, false, bl,
                                                                 (short)0, acc[0][ns], false, false);
            acc[0][ns] = __builtin_amdgcn_wmma_f32_16x16x32_bf16(false, al0, false, bh,
                                                                 (short)0, acc[0][ns], false, false);
            acc[1][ns] = __builtin_amdgcn_wmma_f32_16x16x32_bf16(false, ah1, false, bh,
                                                                 (short)0, acc[1][ns], false, false);
            acc[1][ns] = __builtin_amdgcn_wmma_f32_16x16x32_bf16(false, ah1, false, bl,
                                                                 (short)0, acc[1][ns], false, false);
            acc[1][ns] = __builtin_amdgcn_wmma_f32_16x16x32_bf16(false, al1, false, bh,
                                                                 (short)0, acc[1][ns], false, false);
        }
    }

    // C/D layout: vgpr v, lanes 0-15 -> M=v, lanes 16-31 -> M=v+8
#pragma unroll
    for (int ms = 0; ms < 2; ++ms) {
#pragma unroll
        for (int ns = 0; ns < 4; ++ns) {
            const int n = (colT + ns) * 16 + l15;
#pragma unroll
            for (int v = 0; v < 8; ++v) {
                const int r = (rowT + ms) * 16 + v + 8 * half;
                C[(size_t)r * N + n] = acc[ms][ns][v];
            }
        }
    }
}

// ---------------------------------------------------------------------
// BatchNorm statistics: column sums over B rows, via per-chunk atomics.
// ---------------------------------------------------------------------
__global__ void bn_zero_stats(float* __restrict__ sum, float* __restrict__ sumsq) {
    int f = threadIdx.x;   // 512 threads
    sum[f] = 0.f; sumsq[f] = 0.f;
}

__global__ void bn_stats(const float* __restrict__ H,
                         float* __restrict__ sum, float* __restrict__ sumsq) {
    const int f  = blockIdx.x * 256 + threadIdx.x;   // feature
    const int r0 = blockIdx.y * 256;                 // row chunk
    float s = 0.f, ss = 0.f;
    for (int r = 0; r < 256; ++r) {
        float v = H[(size_t)(r0 + r) * Hh + f];
        s += v; ss += v * v;
    }
    atomicAdd(&sum[f], s);
    atomicAdd(&sumsq[f], ss);
}

// Fold BN into affine: Hbn = H*s + t
__global__ void bn_finalize(const float* __restrict__ sum, const float* __restrict__ sumsq,
                            const float* __restrict__ bn_g, const float* __restrict__ bn_b,
                            float* __restrict__ s, float* __restrict__ t) {
    int f = threadIdx.x;   // 512 threads
    float mean = sum[f] * (1.f / Bn);
    float var  = sumsq[f] * (1.f / Bn) - mean * mean;
    float sc   = bn_g[f] * rsqrtf(var + EPS);
    s[f] = sc;
    t[f] = bn_b[f] - mean * sc;
}

// ---------------------------------------------------------------------
// Fused per-row epilogue: BN apply, LayerNorm+ReLU mask, sigmoid gate,
// concat, residual combine, clamp, row L2-normalize.  1 block = 1 row.
// ---------------------------------------------------------------------
__global__ __launch_bounds__(256) void elem_kernel(
        const float* __restrict__ H, const float* __restrict__ s, const float* __restrict__ t,
        const float* __restrict__ G, const float* __restrict__ gate_b,
        const float* __restrict__ ln_g, const float* __restrict__ ln_b,
        const float* __restrict__ bsv,
        const float* __restrict__ x0, const float* __restrict__ xin,
        float* __restrict__ xout) {
    __shared__ float red[256];
    const int row = blockIdx.x;
    const int tid = threadIdx.x;
    const size_t hoff = (size_t)row * Hh;
    const size_t xoff = (size_t)row * Dd;
    const int f0 = tid, f1 = tid + 256;

    float hbn0 = H[hoff + f0] * s[f0] + t[f0];
    float hbn1 = H[hoff + f1] * s[f1] + t[f1];

    // LayerNorm mean
    red[tid] = hbn0 + hbn1; __syncthreads();
    for (int o = 128; o > 0; o >>= 1) { if (tid < o) red[tid] += red[tid + o]; __syncthreads(); }
    float lmu = red[0] * (1.f / Hh);
    __syncthreads();
    // LayerNorm var (biased)
    float d0 = hbn0 - lmu, d1 = hbn1 - lmu;
    red[tid] = d0 * d0 + d1 * d1; __syncthreads();
    for (int o = 128; o > 0; o >>= 1) { if (tid < o) red[tid] += red[tid + o]; __syncthreads(); }
    float rstd = rsqrtf(red[0] * (1.f / Hh) + EPS);
    __syncthreads();

    float mask0 = fmaxf(d0 * rstd * ln_g[f0] + ln_b[f0], 0.f);
    float mask1 = fmaxf(d1 * rstd * ln_g[f1] + ln_b[f1], 0.f);
    float g0 = 1.f / (1.f + __expf(-(G[hoff + f0] + gate_b[f0])));
    float g1 = 1.f / (1.f + __expf(-(G[hoff + f1] + gate_b[f1])));
    float hg0 = hbn0 * g0, hg1 = hbn1 * g1;

    // Hc = [Hg, Hg*mask]; x = clamp(x0*(Hc+bs)+x); then L2-normalize row
    float hc[4] = { hg0, hg1, hg0 * mask0, hg1 * mask1 };
    int   jj[4] = { f0, f1, Hh + f0, Hh + f1 };
    float xn[4]; float ss2 = 0.f;
#pragma unroll
    for (int q = 0; q < 4; ++q) {
        int j = jj[q];
        float v = x0[xoff + j] * (hc[q] + bsv[j]) + xin[xoff + j];
        v = fminf(fmaxf(v, -10.f), 10.f);
        xn[q] = v; ss2 += v * v;
    }
    red[tid] = ss2; __syncthreads();
    for (int o = 128; o > 0; o >>= 1) { if (tid < o) red[tid] += red[tid + o]; __syncthreads(); }
    float inv = 1.f / (sqrtf(red[0]) + 1e-12f);
#pragma unroll
    for (int q = 0; q < 4; ++q) xout[xoff + jj[q]] = xn[q] * inv;
}

// ---------------------------------------------------------------------
// Final logit: one wave per row, dot(x[row], sfc_w) + sfc_b
// ---------------------------------------------------------------------
__global__ __launch_bounds__(256) void logit_kernel(
        const float* __restrict__ x, const float* __restrict__ w,
        const float* __restrict__ b, float* __restrict__ out) {
    const int lane = threadIdx.x & 31;
    const int wave = threadIdx.x >> 5;
    const int row = blockIdx.x * 8 + wave;
    const size_t off = (size_t)row * Dd;
    float acc = 0.f;
    for (int j = lane; j < Dd; j += 32) acc += x[off + j] * w[j];
#pragma unroll
    for (int o = 16; o > 0; o >>= 1) acc += __shfl_xor(acc, o, 32);
    if (lane == 0) out[row] = acc + b[0];
}

// ---------------------------------------------------------------------
extern "C" void kernel_launch(void* const* d_in, const int* in_sizes, int n_in,
                              void* d_out, int out_size, void* d_ws, size_t ws_size,
                              hipStream_t stream) {
    const float* x_in   = (const float*)d_in[0];
    const float* Ws     = (const float*)d_in[1];   // [L, Hh, Dd]
    const float* bs     = (const float*)d_in[2];   // [L, Dd]
    const float* gate_w = (const float*)d_in[3];   // [L, Hh, Hh]
    const float* gate_b = (const float*)d_in[4];   // [L, Hh]
    const float* ln_g   = (const float*)d_in[5];
    const float* ln_b   = (const float*)d_in[6];
    const float* bn_g   = (const float*)d_in[7];
    const float* bn_b   = (const float*)d_in[8];
    const float* sfc_w  = (const float*)d_in[9];   // [1, Dd]
    const float* sfc_b  = (const float*)d_in[10];  // [1]
    float* out = (float*)d_out;

    // workspace carve-up
    char* ws = (char*)d_ws;
    size_t off = 0;
    auto alloc = [&](size_t bytes) -> void* {
        void* p = ws + off;
        off = (off + bytes + 255) & ~(size_t)255;
        return p;
    };
    float*  x_cur  = (float*)alloc((size_t)Bn * Dd * 4);
    float*  Hbuf   = (float*)alloc((size_t)Bn * Hh * 4);
    float*  Gbuf   = (float*)alloc((size_t)Bn * Hh * 4);
    bf16_t* Xhi    = (bf16_t*)alloc((size_t)Bn * Dd * 2);
    bf16_t* Xlo    = (bf16_t*)alloc((size_t)Bn * Dd * 2);
    bf16_t* HBhi   = (bf16_t*)alloc((size_t)Bn * Hh * 2);
    bf16_t* HBlo   = (bf16_t*)alloc((size_t)Bn * Hh * 2);
    bf16_t* Whi    = (bf16_t*)alloc((size_t)Hh * Dd * 2);
    bf16_t* Wlo    = (bf16_t*)alloc((size_t)Hh * Dd * 2);
    bf16_t* GWhi   = (bf16_t*)alloc((size_t)Hh * Hh * 2);
    bf16_t* GWlo   = (bf16_t*)alloc((size_t)Hh * Hh * 2);
    float*  sumB   = (float*)alloc(Hh * 4);
    float*  sumsqB = (float*)alloc(Hh * 4);
    float*  sAff   = (float*)alloc(Hh * 4);
    float*  tAff   = (float*)alloc(Hh * 4);

    for (int i = 0; i < Ll; ++i) {
        const float* xin = (i == 0) ? x_in : x_cur;

        // swizzle+split operands into fragment-major bf16 hi/lo
        swz_b_split<Dd><<<512, 256, 0, stream>>>(Ws + (size_t)i * Hh * Dd, Whi, Wlo, Hh * Dd);
        swz_b_split<Hh><<<256, 256, 0, stream>>>(gate_w + (size_t)i * Hh * Hh, GWhi, GWlo, Hh * Hh);
        swz_a_split<Dd, false><<<4096, 256, 0, stream>>>(xin, nullptr, nullptr, Xhi, Xlo, Bn * Dd);

        // H = x @ W^T   (split-bf16 WMMA, 32x64 wave tiles)
        gemm_swz<Dd><<<dim3(Hh / 256, Bn / 64), 256, 0, stream>>>(
            Xhi, Xlo, Whi, Wlo, Hbuf, Hh);

        // BatchNorm batch stats -> affine s,t
        bn_zero_stats<<<1, Hh, 0, stream>>>(sumB, sumsqB);
        bn_stats<<<dim3(Hh / 256, Bn / 256), 256, 0, stream>>>(Hbuf, sumB, sumsqB);
        bn_finalize<<<1, Hh, 0, stream>>>(sumB, sumsqB,
                                          bn_g + (size_t)i * Hh, bn_b + (size_t)i * Hh,
                                          sAff, tAff);

        // Hbn (swizzled bf16 hi/lo) for gate GEMM
        swz_a_split<Hh, true><<<4096, 256, 0, stream>>>(Hbuf, sAff, tAff, HBhi, HBlo, Bn * Hh);

        // G = Hbn @ gate_w^T
        gemm_swz<Hh><<<dim3(Hh / 256, Bn / 64), 256, 0, stream>>>(
            HBhi, HBlo, GWhi, GWlo, Gbuf, Hh);

        // fused epilogue (writes x_cur)
        elem_kernel<<<Bn, 256, 0, stream>>>(
            Hbuf, sAff, tAff, Gbuf,
            gate_b + (size_t)i * Hh, ln_g + (size_t)i * Hh, ln_b + (size_t)i * Hh,
            bs + (size_t)i * Dd, x_in, xin, x_cur);
    }

    logit_kernel<<<Bn / 8, 256, 0, stream>>>(x_cur, sfc_w, sfc_b, out);
}
